// GNN_32615981646069
// MI455X (gfx1250) — matrix operand order repaired and checked by reading, not compiled, not run on hardware
//
#include <hip/hip_runtime.h>
#include <hip/hip_bf16.h>
#include <math.h>

typedef __bf16 bf16_t;
typedef bf16_t v16bf __attribute__((ext_vector_type(16)));
typedef bf16_t v8bf  __attribute__((ext_vector_type(8)));
typedef float  v8f   __attribute__((ext_vector_type(8)));
typedef float  v4f   __attribute__((ext_vector_type(4)));

#define HDIM   128
#define SDIM   64
#define BATCH  4096
#define MTILE  16      // batch rows per block
#define NTHREADS 256   // 8 waves (wave32)

// ---------------- WMMA helpers (CDNA5 gfx1250, wave32) ----------------

__device__ __forceinline__ v8f wmma_bf16(v16bf a, v16bf b, v8f c) {
    // D(f32 16x16) = A(bf16 16x32) * B(bf16 32x16) + C
    return __builtin_amdgcn_wmma_f32_16x16x32_bf16(
        false, a, false, b, (short)0, c, false, false);
}

// A-fragment from bf16 LDS tile (row-major, stride HDIM).
// Per-lane A layout: halves[0..7]=K[koff..koff+7], halves[8..15]=K[koff+8..+15].
__device__ __forceinline__ v16bf load_a_lds(const bf16_t* p) {
    v8bf lo = *(const v8bf*)(p);        // ds_load_b128
    v8bf hi = *(const v8bf*)(p + 8);    // ds_load_b128
    v16bf a;
#pragma unroll
    for (int i = 0; i < 8; ++i) { a[i] = lo[i]; a[8 + i] = hi[i]; }
    return a;
}

// B-fragment: weight stored (o, h) row-major == column-major K x N; per lane
// 16 contiguous K values of column n.
__device__ __forceinline__ v16bf load_b(const bf16_t* __restrict__ p) {
    return *(const v16bf*)(p);          // 32B -> 2x global_load_b128
}

// ---------------- weight f32 -> bf16 prologue ----------------

__global__ void cvt_f32_bf16(const float* __restrict__ src,
                             bf16_t* __restrict__ dst, int n) {
    int i = blockIdx.x * blockDim.x + threadIdx.x;
    if (i < n) dst[i] = (bf16_t)src[i];
}

// ---------------- fused state1 -> sum -> state3 -> GRU ----------------

__global__ __launch_bounds__(NTHREADS)
void fused_gnn_gru(const float*  __restrict__ X,     // (B,S,H) f32
                   const bf16_t* __restrict__ W1b,   // (S,H,H) bf16
                   const float*  __restrict__ b1,    // (S,H)
                   const bf16_t* __restrict__ W3b,   // (S,H,H) bf16
                   const float*  __restrict__ b3,    // (S,H)
                   const bf16_t* __restrict__ Wihb,  // (3H,H) bf16
                   const float*  __restrict__ bih,   // (3H)
                   const bf16_t* __restrict__ Whhb,  // (3H,H) bf16
                   const float*  __restrict__ bhh,   // (3H)
                   float* __restrict__ out)          // (B,S,H) f32
{
    __shared__ bf16_t xtile[2][MTILE * HDIM]; // staged X_s tile, bf16, dbl-buffered
    __shared__ bf16_t buf2[MTILE * HDIM];     // state2 tile, bf16
    __shared__ bf16_t buf3[MTILE * HDIM];     // state3 tile, bf16

    const int tid  = threadIdx.x;
    const int wave = tid >> 5;           // 0..7, owns output columns [16w,16w+16)
    const int lane = tid & 31;
    const int lh   = lane >> 4;          // lane half (0/1)
    const int ln   = lane & 15;          // A: row M; B/C: column N
    const int koff = 16 * lh;            // A K-offset for this lane half
    const int kh   = 16 * lh;            // B K-offset for this lane half
    const int b0   = blockIdx.x * MTILE;
    const int n0   = wave * 16;
    const int o    = n0 + ln;            // output column of this lane

    // cooperative staging coords: thread converts 8 consecutive f32 -> bf16
    const int sm = tid >> 4;             // row 0..15
    const int sc = (tid & 15) * 8;       // col 0..120
    const float* stage_src = X + (size_t)(b0 + sm) * (SDIM * HDIM) + sc;

    auto stage = [&](int s, int p) {
        const float* src = stage_src + s * HDIM;
        v4f x0 = *(const v4f*)(src);
        v4f x1 = *(const v4f*)(src + 4);
        v8bf v;
#pragma unroll
        for (int i = 0; i < 4; ++i) { v[i] = (bf16_t)x0[i]; v[4 + i] = (bf16_t)x1[i]; }
        *(v8bf*)(&xtile[p][sm * HDIM + sc]) = v;   // ds_store_b128
    };

    // per-lane base of this lane's weight column slice (64B cacheline head)
    const size_t wcol = (size_t)o * HDIM + kh;

    // ============ Phase 1: total[b,o] = sum_s (X_s @ W1[s]^T + b1[s]) ============
    v8f total = {};
    for (int s = 0; s < SDIM; ++s) {
        stage(s, s & 1);
        // prefetch next slot's streams while this slot computes
        if (s + 1 < SDIM) {
            __builtin_prefetch(W1b + (size_t)(s + 1) * HDIM * HDIM + wcol, 0, 3);
            __builtin_prefetch(stage_src + (s + 1) * HDIM, 0, 3);
        }
        __syncthreads();
        const bf16_t* xa = &xtile[s & 1][ln * HDIM + koff];
        const bf16_t* w  = W1b + (size_t)s * HDIM * HDIM + wcol;
        v8f acc = {};
#pragma unroll
        for (int k0 = 0; k0 < HDIM; k0 += 32)
            acc = wmma_bf16(load_a_lds(xa + k0), load_b(w + k0), acc);
        float bias = b1[s * HDIM + o];
#pragma unroll
        for (int i = 0; i < 8; ++i) total[i] += acc[i] + bias;
    }
    __syncthreads();

    // slot-invariant GRU weight pointers / biases (hoisted)
    const bf16_t* wir = Wihb + (size_t)(0 * HDIM + o) * HDIM + kh;
    const bf16_t* wiz = Wihb + (size_t)(1 * HDIM + o) * HDIM + kh;
    const bf16_t* win = Wihb + (size_t)(2 * HDIM + o) * HDIM + kh;
    const bf16_t* whr = Whhb + (size_t)(0 * HDIM + o) * HDIM + kh;
    const bf16_t* whz = Whhb + (size_t)(1 * HDIM + o) * HDIM + kh;
    const bf16_t* whn = Whhb + (size_t)(2 * HDIM + o) * HDIM + kh;
    const float bir = bih[o], biz = bih[HDIM + o], bin = bih[2 * HDIM + o];
    const float bhr = bhh[o], bhz = bhh[HDIM + o], bhn = bhh[2 * HDIM + o];

    // ============ Phase 2: per slot s, fused state2 -> state3 -> GRU ============
    for (int s = 0; s < SDIM; ++s) {
        const int p = s & 1;
        stage(s, p);
        // prefetch next slot's weight slices across the barrier boundary
        if (s + 1 < SDIM) {
            __builtin_prefetch(W1b + (size_t)(s + 1) * HDIM * HDIM + wcol, 0, 3);
            __builtin_prefetch(W3b + (size_t)(s + 1) * HDIM * HDIM + wcol, 0, 3);
            __builtin_prefetch(stage_src + (s + 1) * HDIM, 0, 3);
        }
        __syncthreads();
        const bf16_t* xa = &xtile[p][ln * HDIM + koff];

        // --- step 1: recompute state1_s; state2 = total - state1_s -> LDS bf16
        {
            const bf16_t* w = W1b + (size_t)s * HDIM * HDIM + wcol;
            v8f acc = {};
#pragma unroll
            for (int k0 = 0; k0 < HDIM; k0 += 32)
                acc = wmma_bf16(load_a_lds(xa + k0), load_b(w + k0), acc);
            float bias = b1[s * HDIM + o];
#pragma unroll
            for (int i = 0; i < 8; ++i) {
                // C element (i, lane): m = i + 8*lh, col = o
                buf2[(i + 8 * lh) * HDIM + o] =
                    (bf16_t)(total[i] - (acc[i] + bias));
            }
        }
        __syncthreads();

        // --- step 2: state3 = state2 @ W3[s]^T + b3[s] -> LDS bf16
        {
            const bf16_t* w = W3b + (size_t)s * HDIM * HDIM + wcol;
            v8f acc = {};
#pragma unroll
            for (int k0 = 0; k0 < HDIM; k0 += 32)
                acc = wmma_bf16(load_a_lds(&buf2[ln * HDIM + koff + k0]),
                                load_b(w + k0), acc);
            float bias = b3[s * HDIM + o];
#pragma unroll
            for (int i = 0; i < 8; ++i)
                buf3[(i + 8 * lh) * HDIM + o] = (bf16_t)(acc[i] + bias);
        }
        __syncthreads();

        // --- step 3: GRU gates; wave w handles hidden columns [16w,16w+16).
        {
            v8f gir = {}, giz = {}, gin = {}, ghr = {}, ghz = {}, ghn = {};
#pragma unroll
            for (int k0 = 0; k0 < HDIM; k0 += 32) {
                v16bf ax = load_a_lds(&buf3[ln * HDIM + koff + k0]); // x = state3
                v16bf ah = load_a_lds(xa + k0);                      // h = X_s
                gir = wmma_bf16(ax, load_b(wir + k0), gir);
                giz = wmma_bf16(ax, load_b(wiz + k0), giz);
                gin = wmma_bf16(ax, load_b(win + k0), gin);
                ghr = wmma_bf16(ah, load_b(whr + k0), ghr);
                ghz = wmma_bf16(ah, load_b(whz + k0), ghz);
                ghn = wmma_bf16(ah, load_b(whn + k0), ghn);
            }
#pragma unroll
            for (int i = 0; i < 8; ++i) {
                float r = 1.0f / (1.0f + __expf(-(gir[i] + bir + ghr[i] + bhr)));
                float z = 1.0f / (1.0f + __expf(-(giz[i] + biz + ghz[i] + bhz)));
                float n = tanhf((gin[i] + bin) + r * (ghn[i] + bhn));
                size_t idx = (size_t)(b0 + i + 8 * lh) * (SDIM * HDIM)
                           + (size_t)s * HDIM + o;
                float h = X[idx];                       // residual in full f32
                out[idx] = h + (1.0f - z) * n + z * h;  // feature_input + h_new
            }
        }
        // next iteration's stage writes the other xtile parity; barriers inside
        // the next iteration cover buf2/buf3 reuse.
    }
}

// ---------------- launch ----------------

extern "C" void kernel_launch(void* const* d_in, const int* in_sizes, int n_in,
                              void* d_out, int out_size, void* d_ws, size_t ws_size,
                              hipStream_t stream) {
    const float* X   = (const float*)d_in[0];
    const float* W1  = (const float*)d_in[1];
    const float* b1  = (const float*)d_in[2];
    const float* W3  = (const float*)d_in[3];
    const float* b3  = (const float*)d_in[4];
    const float* Wih = (const float*)d_in[5];
    const float* bih = (const float*)d_in[6];
    const float* Whh = (const float*)d_in[7];
    const float* bhh = (const float*)d_in[8];
    float* out = (float*)d_out;

    const int nW  = SDIM * HDIM * HDIM;   // 1,048,576 per weight tensor
    const int nWg = 3 * HDIM * HDIM;      // 49,152 per GRU weight

    bf16_t* W1b  = (bf16_t*)d_ws;
    bf16_t* W3b  = W1b + nW;
    bf16_t* Wihb = W3b + nW;
    bf16_t* Whhb = Wihb + nWg;

    cvt_f32_bf16<<<(nW  + 255) / 256, 256, 0, stream>>>(W1,  W1b,  nW);
    cvt_f32_bf16<<<(nW  + 255) / 256, 256, 0, stream>>>(W3,  W3b,  nW);
    cvt_f32_bf16<<<(nWg + 255) / 256, 256, 0, stream>>>(Wih, Wihb, nWg);
    cvt_f32_bf16<<<(nWg + 255) / 256, 256, 0, stream>>>(Whh, Whhb, nWg);

    fused_gnn_gru<<<BATCH / MTILE, NTHREADS, 0, stream>>>(
        X, W1b, b1, W3b, b3, Wihb, bih, Whhb, bhh, out);
}
